// FPSAttn_58514634441159
// MI455X (gfx1250) — matrix operand order repaired and checked by reading, not compiled, not run on hardware
//
#include <hip/hip_runtime.h>
#include <hip/hip_bf16.h>
#include <math.h>

typedef __attribute__((ext_vector_type(16))) _Float16 v16h;
typedef __attribute__((ext_vector_type(8)))  _Float16 h8;
typedef __attribute__((ext_vector_type(2)))  _Float16 h2;
typedef __attribute__((ext_vector_type(8)))  float    v8f;

// ---------------------------------------------------------------------------
// WMMA helpers (CDNA5 wave32, V_WMMA_F32_16X16X32_F16)
// C/D layout: VGPR i, lane l -> row = i + 8*(l>>4), col = l&15
// A layout (16x32 f16): per lane the fragment is rp[8hs..8hs+7] ++ rp[16+8hs..+7]
// B layout (32x16 f16): per lane K = 8hs + 0..7 ++ 16 + 8hs + 0..7, contiguous runs when B is stored
//   n-major (one row per output column n)
// ---------------------------------------------------------------------------
__device__ __forceinline__ v8f wmma_f16(v16h a, v16h b, v8f c) {
    return __builtin_amdgcn_wmma_f32_16x16x32_f16(
        false, a, false, b, (short)0, c, false, false);
}

__device__ __forceinline__ h8 ld8(const _Float16* p) {
    return *(const h8*)p;
}

// A fragment from row-major [*, ld] at (row0, col0); kmaxRel = valid K count
__device__ __forceinline__ v16h ldsA(const _Float16* base, int row0, int col0,
                                     int ld, int kmaxRel, int lane) {
    const int m = lane & 15, hs = lane >> 4;
    const _Float16* rp = base + (row0 + m) * ld + col0;
    h8 lo = ld8(rp + 8 * hs);          // kb = 8hs .. 8hs+7  (< 16, always valid)
    h8 hi;
    if (kmaxRel > 16) hi = ld8(rp + 16 + 8 * hs);
    else              hi = (h8){};
    return __builtin_shufflevector(lo, hi, 0, 1, 2, 3, 4, 5, 6, 7,
                                   8, 9, 10, 11, 12, 13, 14, 15);
}

// A fragment with permuted rows: row = rows[row0+m]
__device__ __forceinline__ v16h ldsA_perm(const _Float16* base, const int* rows,
                                          int row0, int col0, int ld,
                                          int kmaxRel, int lane) {
    const int m = lane & 15, hs = lane >> 4;
    const _Float16* rp = base + rows[row0 + m] * ld + col0;
    h8 lo = ld8(rp + 8 * hs);
    h8 hi;
    if (kmaxRel > 16) hi = ld8(rp + 16 + 8 * hs);
    else              hi = (h8){};
    return __builtin_shufflevector(lo, hi, 0, 1, 2, 3, 4, 5, 6, 7,
                                   8, 9, 10, 11, 12, 13, 14, 15);
}

// B fragment where B[k][n] = src[rown*ld + col0 + k]  (k contiguous, per-lane row)
__device__ __forceinline__ v16h ldsBT(const _Float16* base, int rown, int col0,
                                      int ld, int kmaxRel, int lane) {
    const int hs = lane >> 4;
    const _Float16* rp = base + rown * ld + col0;  // K = 8hs + 0..7 ++ 16 + 8hs + 0..7
    h8 p0, p1;
    if (8 * hs + 7  < kmaxRel) p0 = ld8(rp + 8 * hs);      else p0 = (h8){};
    if (16 + 8 * hs + 7 < kmaxRel) p1 = ld8(rp + 16 + 8 * hs); else p1 = (h8){};
    return __builtin_shufflevector(p0, p1, 0, 1, 2, 3, 4, 5, 6, 7,
                                   8, 9, 10, 11, 12, 13, 14, 15);
}

// ---------------------------------------------------------------------------
// small utility kernels
// ---------------------------------------------------------------------------
__global__ void f2h_kernel(const float* __restrict__ s, _Float16* __restrict__ d, int n) {
    int i = blockIdx.x * blockDim.x + threadIdx.x;
    if (i < n) d[i] = (_Float16)s[i];
}

// convert + transpose: src is (K x N) row-major, dst is (N x K) row-major
__global__ void f2hT_kernel(const float* __restrict__ s, _Float16* __restrict__ d,
                            int K, int N) {
    int i = blockIdx.x * blockDim.x + threadIdx.x;
    if (i < K * N) {
        int n = i / K, k = i % K;
        d[n * K + k] = (_Float16)s[k * N + n];
    }
}

__global__ void zero_kernel(float* __restrict__ p, int n) {
    int i = blockIdx.x * blockDim.x + threadIdx.x;
    if (i < n) p[i] = 0.f;
}

// ---------------------------------------------------------------------------
// Attention kernel: one block (8 waves) per 8x8 patch sequence (N=64, c=64)
// Dynamic LDS: xpH[64*64] qH/kH[64*192] vT[192*64] Pst[8*16*64] oBuf[64*192]
//              projS[512] lseS[512] normS[256] mxS[4] posS[512] invS[512]
// (qH region is reused as float spaFT staging after the attention GEMMs)
// ---------------------------------------------------------------------------
#define ATTN_LDS_BYTES 132112

__global__ __launch_bounds__(256) void fps_attn_kernel(
    const float* __restrict__ x,
    const _Float16* __restrict__ WqT, const _Float16* __restrict__ WkT,
    const _Float16* __restrict__ WvT, const _Float16* __restrict__ WoT,
    const float* __restrict__ alpha, const float* __restrict__ beta,
    const float* __restrict__ dw_w, float* __restrict__ out)
{
    extern __shared__ char smem[];
    _Float16* xpH  = (_Float16*)smem;          // 4096
    _Float16* qH   = xpH + 64 * 64;            // 12288 (token-major)
    _Float16* kH   = qH + 64 * 192;            // 12288 (token-major)
    _Float16* vT   = kH + 64 * 192;            // 12288 (d-major: vT[d][token])
    _Float16* Pst  = vT + 192 * 64;            // 8192
    _Float16* oBuf = Pst + 8 * 16 * 64;        // 12288
    float* projS = (float*)(oBuf + 64 * 192);  // 512
    float* lseS  = projS + 512;                // 512
    float* normS = lseS + 512;                 // 256
    float* mxS   = normS + 256;                // 4
    int*   posS  = (int*)(mxS + 4);            // 512
    int*   invS  = posS + 512;                 // 512

    const int tid = threadIdx.x, lane = tid & 31, wave = tid >> 5;
    const int hs = lane >> 4, nn = lane & 15;
    const int p = blockIdx.x;
    const int bb   = p >> 10;
    const int nh_i = (p >> 5) & 31;
    const int nw_i = p & 31;
    const long xbase = (long)bb * 64 * 65536;

    // ---- load patch: xp[t][ch] = x[bb][ch][nh_i*8 + t/8][nw_i*8 + t%8]
    for (int idx = tid; idx < 4096; idx += 256) {
        int t = idx >> 6, ch = idx & 63;
        int gh = nh_i * 8 + (t >> 3), gw = nw_i * 8 + (t & 7);
        xpH[t * 64 + ch] = (_Float16)x[xbase + (long)ch * 65536 + gh * 256 + gw];
    }
    __syncthreads();

    // ---- hashing: squared norms per (token, head)
    if (tid < 256) {
        int t = tid >> 2, h = tid & 3;
        float s = 0.f;
        for (int j = 0; j < 16; ++j) {
            float v = (float)xpH[t * 64 + h * 16 + j];
            s += v * v;
        }
        normS[tid] = s;
    }
    __syncthreads();
    if (tid < 4) {
        float mx = 0.f;
        for (int t = 0; t < 64; ++t) mx = fmaxf(mx, normS[t * 4 + tid]);
        mxS[tid] = mx;
    }
    __syncthreads();
    if (tid < 256) {
        int t = tid >> 2, h = tid & 3;
        float ext = sqrtf(fmaxf(mxS[h] - normS[tid], 0.f));
        for (int r = 0; r < 2; ++r) {
            float s = beta[r];
            for (int j = 0; j < 16; ++j)
                s += (float)xpH[t * 64 + h * 16 + j] * alpha[j * 2 + r];
            s += ext * alpha[16 * 2 + r];   // alpha[17] multiplies zero
            projS[(t * 4 + h) * 2 + r] = s;
        }
    }
    __syncthreads();
    // stable argsort via rank counting (N=64)
    for (int idx = tid; idx < 512; idx += 256) {
        int r = idx >> 8, h = (idx >> 6) & 3, t = idx & 63;
        float pt = projS[(t * 4 + h) * 2 + r];
        int rank = 0;
        for (int j = 0; j < 64; ++j) {
            float pj = projS[(j * 4 + h) * 2 + r];
            rank += (pj < pt) || (pj == pt && j < t);
        }
        posS[(r * 4 + h) * 64 + rank] = t;
        invS[(r * 4 + h) * 64 + t]    = rank;
    }
    __syncthreads();

    // ---- QKV projection: 3 x (64x64 @ 64x192); B from pre-transposed weights
    for (int tile = wave; tile < 144; tile += 8) {
        int mat = tile / 48, rem = tile % 48;
        int mt = rem / 12, ct = rem % 12;
        const _Float16* WT = (mat == 0) ? WqT : (mat == 1) ? WkT : WvT;
        v8f acc = {};
#pragma unroll
        for (int kk = 0; kk < 2; ++kk) {
            v16h a = ldsA(xpH, mt * 16, kk * 32, 64, 32, lane);
            v16h b = ldsBT(WT, ct * 16 + nn, kk * 32, 64, 32, lane);
            acc = wmma_f16(a, b, acc);
        }
        if (mat == 2) {
            // V stored transposed: vT[dglobal][token]; per lane the 8 C rows
            // are contiguous tokens -> one packed b128 store
            h8 pk;
#pragma unroll
            for (int i = 0; i < 8; ++i) pk[i] = (_Float16)acc[i];
            *(h8*)(vT + (ct * 16 + nn) * 64 + mt * 16 + 8 * hs) = pk;
        } else {
            _Float16* dst = (mat == 0) ? qH : kH;
#pragma unroll
            for (int i = 0; i < 8; ++i)
                dst[(mt * 16 + i + 8 * hs) * 192 + ct * 16 + nn] = (_Float16)acc[i];
        }
    }
    __syncthreads();

    // ---- per-wave attention: wave -> (round r, head h)
    const int r = wave >> 2, h = wave & 3;
    const int* pp = posS + (r * 4 + h) * 64;
    _Float16* Pw = Pst + wave * 1024;   // 16x64 staging for current row tile

    // per-lane original tokens for S columns / P scatter columns
    int tok[4];
#pragma unroll
    for (int ct = 0; ct < 4; ++ct) tok[ct] = pp[ct * 16 + nn];

    v8f bo[4][3];
    {
        v8f z = {};
#pragma unroll
        for (int mt = 0; mt < 4; ++mt)
#pragma unroll
            for (int c2 = 0; c2 < 3; ++c2) bo[mt][c2] = z;
    }

#pragma unroll
    for (int mt = 0; mt < 4; ++mt) {
        // S row stripe: (16 x 64), K = d = 48 (zero-padded to 64)
        v8f s4[4] = {{}, {}, {}, {}};
#pragma unroll
        for (int ct = 0; ct < 4; ++ct) {
#pragma unroll
            for (int kk = 0; kk < 2; ++kk) {
                int kRel = (kk == 0) ? 32 : 16;
                v16h a = ldsA_perm(qH, pp, mt * 16, h * 48 + kk * 32, 192, kRel, lane);
                v16h b = ldsBT(kH, tok[ct], h * 48 + kk * 32, 192, kRel, lane);
                s4[ct] = wmma_f16(a, b, s4[ct]);
            }
        }
        // row softmax; scatter P into ORIGINAL-token columns so the second
        // GEMM contracts against unpermuted V
#pragma unroll
        for (int i = 0; i < 8; ++i) {
            float m = fmaxf(fmaxf(s4[0][i], s4[1][i]), fmaxf(s4[2][i], s4[3][i]));
            for (int msk = 1; msk < 16; msk <<= 1) m = fmaxf(m, __shfl_xor(m, msk, 32));
            float e0 = expf(s4[0][i] - m), e1 = expf(s4[1][i] - m);
            float e2 = expf(s4[2][i] - m), e3 = expf(s4[3][i] - m);
            float sum = e0 + e1 + e2 + e3;
            for (int msk = 1; msk < 16; msk <<= 1) sum += __shfl_xor(sum, msk, 32);
            float inv = 1.f / sum;
            int lrow = i + 8 * hs;
            Pw[lrow * 64 + tok[0]] = (_Float16)(e0 * inv);
            Pw[lrow * 64 + tok[1]] = (_Float16)(e1 * inv);
            Pw[lrow * 64 + tok[2]] = (_Float16)(e2 * inv);
            Pw[lrow * 64 + tok[3]] = (_Float16)(e3 * inv);
            if (nn == 0) lseS[(r * 4 + h) * 64 + mt * 16 + lrow] = m + logf(sum);
        }
        // bo row stripe = P(16x64, original-token cols) @ V(64x48) via vT
#pragma unroll
        for (int c2 = 0; c2 < 3; ++c2) {
            v8f acc = bo[mt][c2];
#pragma unroll
            for (int kk = 0; kk < 2; ++kk) {
                v16h a = ldsA(Pw, 0, kk * 32, 64, 32, lane);
                v16h b = ldsBT(vT, h * 48 + c2 * 16 + nn, kk * 32, 64, 32, lane);
                acc = wmma_f16(a, b, acc);
            }
            bo[mt][c2] = acc;
        }
    }
    __syncthreads();   // all lse written

    // ---- round-weighted scatter back (round 0 writes, round 1 adds)
    for (int phase = 0; phase < 2; ++phase) {
        if (r == phase) {
#pragma unroll
            for (int mt = 0; mt < 4; ++mt) {
#pragma unroll
                for (int i = 0; i < 8; ++i) {
                    int m = mt * 16 + i + 8 * hs;
                    int t = pp[m];
                    float l0 = lseS[(0 * 4 + h) * 64 + invS[(0 * 4 + h) * 64 + t]];
                    float l1 = lseS[(1 * 4 + h) * 64 + invS[(1 * 4 + h) * 64 + t]];
                    float mx = fmaxf(l0, l1);
                    float z = expf(l0 - mx) + expf(l1 - mx);
                    float lr = (r == 0) ? l0 : l1;
                    float prob = expf(lr - mx) / z;
#pragma unroll
                    for (int c2 = 0; c2 < 3; ++c2) {
                        int col = h * 48 + c2 * 16 + nn;
                        float val = bo[mt][c2][i] * prob;
                        if (phase == 0)
                            oBuf[t * 192 + col] = (_Float16)val;
                        else
                            oBuf[t * 192 + col] =
                                (_Float16)((float)oBuf[t * 192 + col] + val);
                    }
                }
            }
        }
        __syncthreads();
    }

    // ---- output projection (64x192 @ 192x64); stage spa*dw0 transposed in LDS
    float* spaFT = (float*)qH;   // spaFT[ch][t], 16 KB (qH/kH are dead)
    for (int tile = wave; tile < 16; tile += 8) {
        int mt = tile >> 2, ct = tile & 3;
        v8f acc = {};
#pragma unroll
        for (int kk = 0; kk < 6; ++kk) {
            v16h a = ldsA(oBuf, mt * 16, kk * 32, 192, 32, lane);
            v16h b = ldsBT(WoT, ct * 16 + nn, kk * 32, 192, 32, lane);
            acc = wmma_f16(a, b, acc);
        }
        int ch = ct * 16 + nn;
        float dw0 = dw_w[ch * 2 + 0];
        float* dstp = spaFT + ch * 64 + mt * 16 + 8 * hs;  // 8 contiguous tokens
        float4 lo = make_float4(acc[0] * dw0, acc[1] * dw0, acc[2] * dw0, acc[3] * dw0);
        float4 hi = make_float4(acc[4] * dw0, acc[5] * dw0, acc[6] * dw0, acc[7] * dw0);
        *(float4*)dstp = lo;
        *(float4*)(dstp + 4) = hi;
    }
    __syncthreads();

    // ---- coalesced output: float4 along the fast spatial axis
    for (int idx = tid; idx < 1024; idx += 256) {
        int ch = idx >> 4;
        int t0 = (idx & 15) * 4;
        int gh = nh_i * 8 + (t0 >> 3), gw = nw_i * 8 + (t0 & 7);
        float4 vv = *(const float4*)(spaFT + ch * 64 + t0);
        *(float4*)(out + xbase + (long)ch * 65536 + gh * 256 + gw) = vv;
    }
}

// ---------------------------------------------------------------------------
// FMAM part 1: ctx numerator + esum over 128-column chunks of n
// LDS: Xc[64][128] (c-major, A of ctx GEMM), XcT[128][64] (j-major, B of yk),
//      Ec[64][128]
// ---------------------------------------------------------------------------
#define C1_LDS_BYTES (3 * 64 * 128 * 2)

__global__ __launch_bounds__(256) void fmam_ctx_kernel(
    const float* __restrict__ x, const _Float16* __restrict__ fmamH,
    const float* __restrict__ fmam_b,
    float* __restrict__ ctx_num, float* __restrict__ esum)
{
    extern __shared__ char smem[];
    _Float16* Xc  = (_Float16*)smem;     // 64 x 128
    _Float16* XcT = Xc + 64 * 128;       // 128 x 64
    _Float16* Ec  = XcT + 128 * 64;      // 64 x 128
    const int tid = threadIdx.x, lane = tid & 31, wave = tid >> 5;
    const int hs = lane >> 4, nn = lane & 15;
    const int bb = blockIdx.x >> 9;
    const int n0 = (blockIdx.x & 511) * 128;
    const long xb = (long)bb * 64 * 65536;

    // vectorized stage: float2 global loads -> Xc (packed) + XcT (transposed)
    for (int idx = tid; idx < 4096; idx += 256) {
        int ch = idx >> 6, j2 = (idx & 63) * 2;
        const float* xp = x + xb + (long)ch * 65536 + n0 + j2;
        __builtin_prefetch(xp + 128, 0, 1);          // global_prefetch_b8
        float2 vv = *(const float2*)xp;
        h2 o; o[0] = (_Float16)vv.x; o[1] = (_Float16)vv.y;
        *(h2*)(Xc + ch * 128 + j2) = o;
        XcT[(j2 + 0) * 64 + ch] = o[0];
        XcT[(j2 + 1) * 64 + ch] = o[1];
    }
    __syncthreads();

    // E = exp(fmam_w @ X + b), 4x8 tiles; B contiguous via XcT
    for (int tile = wave; tile < 32; tile += 8) {
        int mt = tile >> 3, nt = tile & 7;
        v8f acc = {};
#pragma unroll
        for (int kk = 0; kk < 2; ++kk) {
            v16h a = ldsA(fmamH, mt * 16, kk * 32, 64, 32, lane);
            v16h b = ldsBT(XcT, nt * 16 + nn, kk * 32, 64, 32, lane);
            acc = wmma_f16(a, b, acc);
        }
#pragma unroll
        for (int i = 0; i < 8; ++i) {
            int o = mt * 16 + i + 8 * hs;
            Ec[o * 128 + nt * 16 + nn] = (_Float16)expf(acc[i] + fmam_b[o]);
        }
    }
    __syncthreads();

    if (tid < 64) {
        float s = 0.f;
        for (int j = 0; j < 128; ++j) s += (float)Ec[tid * 128 + j];
        atomicAdd(&esum[bb * 64 + tid], s);
    }

    // ctx_num += X (64xK) @ E^T (Kx64), K = 128 chunk
    for (int tile = wave; tile < 16; tile += 8) {
        int mt = tile >> 2, nt = tile & 3;
        v8f acc = {};
#pragma unroll
        for (int kk = 0; kk < 4; ++kk) {
            v16h a = ldsA(Xc, mt * 16, kk * 32, 128, 32, lane);
            v16h b = ldsBT(Ec, nt * 16 + nn, kk * 32, 128, 32, lane);
            acc = wmma_f16(a, b, acc);
        }
#pragma unroll
        for (int i = 0; i < 8; ++i) {
            int c = mt * 16 + i + 8 * hs, o = nt * 16 + nn;
            atomicAdd(&ctx_num[bb * 4096 + c * 64 + o], acc[i]);
        }
    }
}

// ---------------------------------------------------------------------------
// FMAM part 2: queries softmax + freq = ctx^T @ q_hat, out += freq * dw_w[:,1]
// LDS: XcT[128][64] (B of yq), Qc[64][128] (raw yq), QcT[128][64] (B of freq),
//      ctxT[64][64]
// ---------------------------------------------------------------------------
#define C2_LDS_BYTES ((3 * 64 * 128 + 64 * 64) * 2)

__global__ __launch_bounds__(256) void fmam_freq_kernel(
    const float* __restrict__ x, const _Float16* __restrict__ qwH,
    const float* __restrict__ q_b, const float* __restrict__ ctx_num,
    const float* __restrict__ esum, const float* __restrict__ dw_w,
    float* __restrict__ out)
{
    extern __shared__ char smem[];
    _Float16* XcT  = (_Float16*)smem;    // 128 x 64
    _Float16* Qc   = XcT + 128 * 64;     // 64 x 128
    _Float16* QcT  = Qc + 64 * 128;      // 128 x 64
    _Float16* ctxT = QcT + 128 * 64;     // 64 x 64: ctxT[k][c] = ctx[c][k]
    const int tid = threadIdx.x, lane = tid & 31, wave = tid >> 5;
    const int hs = lane >> 4, nn = lane & 15;
    const int bb = blockIdx.x >> 9;
    const int n0 = (blockIdx.x & 511) * 128;
    const long xb = (long)bb * 64 * 65536;

    for (int idx = tid; idx < 4096; idx += 256) {
        int ch = idx >> 6, j2 = (idx & 63) * 2;
        const float* xp = x + xb + (long)ch * 65536 + n0 + j2;
        __builtin_prefetch(xp + 128, 0, 1);
        float2 vv = *(const float2*)xp;
        XcT[(j2 + 0) * 64 + ch] = (_Float16)vv.x;
        XcT[(j2 + 1) * 64 + ch] = (_Float16)vv.y;
    }
    for (int idx = tid; idx < 2048; idx += 256) {
        int k = idx >> 5, c2 = (idx & 31) * 2;
        float inv = 1.f / esum[bb * 64 + k];
        h2 o;
        o[0] = (_Float16)(ctx_num[bb * 4096 + (c2 + 0) * 64 + k] * inv);
        o[1] = (_Float16)(ctx_num[bb * 4096 + (c2 + 1) * 64 + k] * inv);
        *(h2*)(ctxT + k * 64 + c2) = o;
    }
    __syncthreads();

    // yq = q_w @ X + q_b (raw, pre-softmax)
    for (int tile = wave; tile < 32; tile += 8) {
        int mt = tile >> 3, nt = tile & 7;
        v8f acc = {};
#pragma unroll
        for (int kk = 0; kk < 2; ++kk) {
            v16h a = ldsA(qwH, mt * 16, kk * 32, 64, 32, lane);
            v16h b = ldsBT(XcT, nt * 16 + nn, kk * 32, 64, 32, lane);
            acc = wmma_f16(a, b, acc);
        }
#pragma unroll
        for (int i = 0; i < 8; ++i) {
            int o = mt * 16 + i + 8 * hs;
            Qc[o * 128 + nt * 16 + nn] = (_Float16)(acc[i] + q_b[o]);
        }
    }
    __syncthreads();

    // column-wise softmax over the 64 channel outputs; write transposed+packed
    if (tid < 128) {
        int j = tid;
        float m = -1e30f;
        for (int o = 0; o < 64; ++o) m = fmaxf(m, (float)Qc[o * 128 + j]);
        float s = 0.f;
        for (int o = 0; o < 64; ++o) s += expf((float)Qc[o * 128 + j] - m);
        float inv = 1.f / s;
        for (int o = 0; o < 64; o += 2) {
            h2 pq;
            pq[0] = (_Float16)(expf((float)Qc[(o + 0) * 128 + j] - m) * inv);
            pq[1] = (_Float16)(expf((float)Qc[(o + 1) * 128 + j] - m) * inv);
            *(h2*)(QcT + j * 64 + o) = pq;
        }
    }
    __syncthreads();

    // freq = ctxT (64x64) @ Qhat (64x128); out += freq * dw1
    for (int tile = wave; tile < 32; tile += 8) {
        int mt = tile >> 3, nt = tile & 7;
        v8f acc = {};
#pragma unroll
        for (int kk = 0; kk < 2; ++kk) {
            v16h a = ldsA(ctxT, mt * 16, kk * 32, 64, 32, lane);
            v16h b = ldsBT(QcT, nt * 16 + nn, kk * 32, 64, 32, lane);
            acc = wmma_f16(a, b, acc);
        }
#pragma unroll
        for (int i = 0; i < 8; ++i) {
            int k = mt * 16 + i + 8 * hs, j = nt * 16 + nn;
            long oidx = xb + (long)k * 65536 + n0 + j;
            out[oidx] = out[oidx] + acc[i] * dw_w[k * 2 + 1];
        }
    }
}

// ---------------------------------------------------------------------------
// launcher
// ---------------------------------------------------------------------------
extern "C" void kernel_launch(void* const* d_in, const int* in_sizes, int n_in,
                              void* d_out, int out_size, void* d_ws, size_t ws_size,
                              hipStream_t stream)
{
    (void)in_sizes; (void)n_in; (void)out_size; (void)ws_size;
    const float* x      = (const float*)d_in[0];
    const float* Wq     = (const float*)d_in[1];
    const float* Wk     = (const float*)d_in[2];
    const float* Wv     = (const float*)d_in[3];
    const float* Wo     = (const float*)d_in[4];
    const float* alpha  = (const float*)d_in[5];
    const float* beta   = (const float*)d_in[6];
    const float* q_w    = (const float*)d_in[7];
    const float* q_b    = (const float*)d_in[8];
    const float* fmam_w = (const float*)d_in[9];
    const float* fmam_b = (const float*)d_in[10];
    const float* dw_w   = (const float*)d_in[11];
    float* out = (float*)d_out;

    // workspace layout (transposed f16 weights)
    _Float16* WqT   = (_Float16*)d_ws;       // 192 x 64
    _Float16* WkT   = WqT + 12288;           // 192 x 64
    _Float16* WvT   = WkT + 12288;           // 192 x 64
    _Float16* WoT   = WvT + 12288;           // 64 x 192
    _Float16* fmamH = WoT + 12288;           // 64 x 64 (A operand, row-major)
    _Float16* qwH   = fmamH + 4096;          // 64 x 64 (A operand, row-major)
    float* ctx_num  = (float*)(qwH + 4096);  // 2*64*64
    float* esum     = ctx_num + 8192;        // 2*64

    f2hT_kernel<<<48, 256, 0, stream>>>(Wq, WqT, 64, 192);
    f2hT_kernel<<<48, 256, 0, stream>>>(Wk, WkT, 64, 192);
    f2hT_kernel<<<48, 256, 0, stream>>>(Wv, WvT, 64, 192);
    f2hT_kernel<<<48, 256, 0, stream>>>(Wo, WoT, 192, 64);
    f2h_kernel<<<16, 256, 0, stream>>>(fmam_w, fmamH, 4096);
    f2h_kernel<<<16, 256, 0, stream>>>(q_w, qwH, 4096);
    zero_kernel<<<33, 256, 0, stream>>>(ctx_num, 8192 + 128);

    fps_attn_kernel<<<2048, 256, ATTN_LDS_BYTES, stream>>>(
        x, WqT, WkT, WvT, WoT, alpha, beta, dw_w, out);

    fmam_ctx_kernel<<<1024, 256, C1_LDS_BYTES, stream>>>(
        x, fmamH, fmam_b, ctx_num, esum);

    fmam_freq_kernel<<<1024, 256, C2_LDS_BYTES, stream>>>(
        x, qwH, q_b, ctx_num, esum, dw_w, out);
}
